// HybridPointNet_65000035058614
// MI455X (gfx1250) — compile-verified, matrix operand order checked by reading, not run on hardware
//
#include <hip/hip_runtime.h>

typedef __attribute__((ext_vector_type(16))) _Float16 v16h;
typedef __attribute__((ext_vector_type(8)))  float    v8f;
typedef __attribute__((ext_vector_type(4)))  _Float16 v4h;
typedef _Float16 f16;

#define EPSBN 1e-5f

// ---------------------------------------------------------------------------
// Fragment helpers (16-bit A/B layout per CDNA5 ISA):
// lane<16: row/col = lane, k = {koff..koff+7, koff+16..koff+23}, koff=0
// lane>=16: same row/col (lane-16), koff=8
// Data laid out so those k's are contiguous: A tiles row-major [m][k],
// B pre-transposed [n][Kp].
// ---------------------------------------------------------------------------
__device__ __forceinline__ v16h frag_load16(const f16* p0) {
  v16h a;
  *((uint4*)&a)     = *(const uint4*)(p0);
  *(((uint4*)&a)+1) = *(const uint4*)(p0 + 16);
  return a;
}
__device__ __forceinline__ void cvt_store4(f16* dst, float4 f) {
  v4h h;
  h[0] = (f16)f.x; h[1] = (f16)f.y; h[2] = (f16)f.z; h[3] = (f16)f.w;
  *(v4h*)dst = h;
}

// ---------------------------------------------------------------------------
// Weight prep: f32 [K][N] -> f16 transposed [Np][Kp]; K padded to 32,
// N padded to 128 (GEMM tile width) so B-fragment loads need no guards.
// ---------------------------------------------------------------------------
__global__ void k_convert_wt(const float* __restrict__ W, int K, int N,
                             int Kp, int Np, f16* __restrict__ Wt) {
  long t = (long)blockIdx.x * 256 + threadIdx.x;
  if (t >= (long)Np * Kp) return;
  int n = (int)(t / Kp), kp = (int)(t % Kp);
  float v = (kp < K && n < N) ? W[(long)kp * N + n] : 0.f;
  Wt[t] = (f16)v;
}

// ---------------------------------------------------------------------------
// Generic WMMA GEMM: Out[M x N] = act(bn((A[M x K] @ W) + bias))
// A: f32 row-major (lda), W given as f16 transposed [Np][Kp] (N padded to
// 128 -> unconditional B loads). Block 256 thr = 8 waves; tile 128x128;
// waves 4(M) x 2(N); 2x4 frags/wave. Guard-free fast k-loop when K%32==0
// and the M-tile is full (all layers except K=3); guarded loop otherwise.
// ---------------------------------------------------------------------------
__global__ void __launch_bounds__(256)
k_gemm(const float* __restrict__ A, int lda, int M, int K,
       const f16* __restrict__ Wt, int N,
       const float* __restrict__ bias, const float* __restrict__ bng,
       const float* __restrict__ bnb, int relu,
       float* __restrict__ Out, long ldo, long ooff) {
  __shared__ __align__(16) f16 Atile[128 * 40];   // 32 k-cols + 8 pad halves
  const int tid  = threadIdx.x;
  const int lane = tid & 31, wave = tid >> 5;
  const int mw = wave & 3, nw = wave >> 2;
  const int m0 = blockIdx.y * 128, n0 = blockIdx.x * 128;
  const int Kp = (K + 31) & ~31;
  const int koff = (lane >> 4) << 3;

  v8f acc[2][4];
  #pragma unroll
  for (int i = 0; i < 2; ++i)
    #pragma unroll
    for (int j = 0; j < 4; ++j)
      #pragma unroll
      for (int e = 0; e < 8; ++e) acc[i][j][e] = 0.f;

  const int rs = tid >> 1, hs = tid & 1;           // staging row / 16-chunk
  const long rowS  = m0 + rs;
  const long rowCl = (rowS < M) ? rowS : (M - 1);
  const float* apS = A + rowCl * (long)lda;
  f16* dst = &Atile[rs * 40 + hs * 16];
  const f16* wBase = Wt + (long)(n0 + nw * 64 + (lane & 15)) * Kp + koff;

  if (((K & 31) == 0) && (m0 + 128 <= M)) {
    // ---------------- fast loop: no bounds checks at all --------------------
    for (int k0 = 0; k0 < K; k0 += 32) {
      {
        const float4* s = (const float4*)(apS + k0 + hs * 16);
        #pragma unroll
        for (int u = 0; u < 4; ++u) cvt_store4(dst + u * 4, s[u]);
        __builtin_prefetch(apS + k0 + 32 + hs * 16, 0, 3);  // near-cache prefetch
      }
      __syncthreads();
      v16h bf[4];
      #pragma unroll
      for (int nt = 0; nt < 4; ++nt)
        bf[nt] = frag_load16(wBase + (long)nt * 16 * Kp + k0);
      #pragma unroll
      for (int mt = 0; mt < 2; ++mt) {
        int rbase = mw * 32 + mt * 16 + (lane & 15);
        v16h af = frag_load16(&Atile[rbase * 40 + koff]);
        #pragma unroll
        for (int nt = 0; nt < 4; ++nt)
          acc[mt][nt] = __builtin_amdgcn_wmma_f32_16x16x32_f16(
              false, af, false, bf[nt], (short)0, acc[mt][nt], false, false);
      }
      __syncthreads();
    }
  } else {
    // ---------------- guarded loop (K=3 layers / ragged tiles) --------------
    for (int k0 = 0; k0 < Kp; k0 += 32) {
      {
        const int kc = k0 + hs * 16;
        #pragma unroll
        for (int u = 0; u < 16; ++u) {
          int k  = kc + u;
          int kk = (k < K) ? k : 0;              // clamp addr, mask value
          float v = apS[kk];
          v = (k < K && rowS < M) ? v : 0.f;
          dst[u] = (f16)v;
        }
      }
      __syncthreads();
      v16h bf[4];
      #pragma unroll
      for (int nt = 0; nt < 4; ++nt)
        bf[nt] = frag_load16(wBase + (long)nt * 16 * Kp + k0);
      #pragma unroll
      for (int mt = 0; mt < 2; ++mt) {
        int rbase = mw * 32 + mt * 16 + (lane & 15);
        v16h af = frag_load16(&Atile[rbase * 40 + koff]);
        #pragma unroll
        for (int nt = 0; nt < 4; ++nt)
          acc[mt][nt] = __builtin_amdgcn_wmma_f32_16x16x32_f16(
              false, af, false, bf[nt], (short)0, acc[mt][nt], false, false);
      }
      __syncthreads();
    }
  }

  // ---- epilogue: bias + BN(eval) + ReLU (stores guarded by true N/M) ----
  #pragma unroll
  for (int mt = 0; mt < 2; ++mt) {
    #pragma unroll
    for (int nt = 0; nt < 4; ++nt) {
      int col = n0 + nw * 64 + nt * 16 + (lane & 15);
      if (col >= N) continue;
      float bv = bias ? bias[col] : 0.f;
      float sc = 1.f, sh = 0.f;
      if (bng) { sc = bng[col] * rsqrtf(1.f + EPSBN); sh = bnb[col]; }
      #pragma unroll
      for (int v = 0; v < 8; ++v) {
        long row = m0 + mw * 32 + mt * 16 + v + ((lane >= 16) ? 8 : 0);
        if (row < M) {
          float y = acc[mt][nt][v] + bv;
          if (bng) y = y * sc + sh;
          if (relu) y = fmaxf(y, 0.f);
          Out[row * ldo + ooff + col] = y;
        }
      }
    }
  }
}

// ---------------------------------------------------------------------------
// Fused DynamicEdgeConv: per block 2 points x 32 padded neighbors (M=64).
// Edge features [x_i, x_j - x_i] built on the fly into LDS (f16, float4
// gathers); l1 WMMA + BN + ReLU into LDS; l2 WMMA; max over k in registers
// + lane16 shuffle. Waves: 2(M: one point each) x 4(N: COUT/4 cols).
// ---------------------------------------------------------------------------
template <int CIN, int COUT>
__global__ void __launch_bounds__(256)
k_edgeconv(const float* __restrict__ X, long ldx, long xoff,
           const int* __restrict__ IDX,
           const f16* __restrict__ W1t, const float* __restrict__ b1,
           const float* __restrict__ bng, const float* __restrict__ bnb,
           const f16* __restrict__ W2t, const float* __restrict__ b2,
           float* __restrict__ Out, long ldo, long ooff) {
  constexpr int K1  = 2 * CIN;
  constexpr int NTW = COUT / 64;     // N-tiles per wave
  constexpr int CQ  = K1 / 4;        // cols per staging thread (= CIN/2)
  __shared__ __align__(16) f16 Aedge[64 * K1];
  __shared__ __align__(16) f16 H1[64 * COUT];

  const int tid = threadIdx.x, lane = tid & 31, wave = tid >> 5;
  const int mw = wave & 1, nw = wave >> 1;
  const long iBase = (long)blockIdx.x * 2;
  const int koff = (lane >> 4) << 3;

  // Phase 1: stage edge features (vectorized)
  {
    const int r = tid >> 2, q = tid & 3;
    const int p = r >> 5, e = r & 31;
    const long i = iBase + p;
    const long j = IDX[i * 30 + (e < 30 ? e : 0)];   // pad = dup neighbor 0
    const float* xi = X + i * ldx + xoff;
    const float* xj = X + j * ldx + xoff;
    f16* dst = &Aedge[r * K1 + q * CQ];
    if (q < 2) {                       // x_i half
      const float4* s = (const float4*)(xi + q * CQ);
      #pragma unroll
      for (int u = 0; u < CQ / 4; ++u) cvt_store4(dst + u * 4, s[u]);
    } else {                           // (x_j - x_i) half
      const float4* si = (const float4*)(xi + (q - 2) * CQ);
      const float4* sj = (const float4*)(xj + (q - 2) * CQ);
      #pragma unroll
      for (int u = 0; u < CQ / 4; ++u) {
        float4 a = si[u], b = sj[u];
        float4 d; d.x = b.x - a.x; d.y = b.y - a.y;
        d.z = b.z - a.z; d.w = b.w - a.w;
        cvt_store4(dst + u * 4, d);
      }
    }
  }
  __syncthreads();

  v8f acc[2][NTW];
  #pragma unroll
  for (int i = 0; i < 2; ++i)
    #pragma unroll
    for (int j = 0; j < NTW; ++j)
      #pragma unroll
      for (int e = 0; e < 8; ++e) acc[i][j][e] = 0.f;

  // Phase 2: l1 = edge @ W1
  for (int k0 = 0; k0 < K1; k0 += 32) {
    v16h bf[NTW];
    #pragma unroll
    for (int nt = 0; nt < NTW; ++nt) {
      int colBase = nw * (COUT / 4) + nt * 16;
      bf[nt] = frag_load16(W1t + (long)(colBase + (lane & 15)) * K1 + k0 + koff);
    }
    #pragma unroll
    for (int mt = 0; mt < 2; ++mt) {
      int rbase = mw * 32 + mt * 16 + (lane & 15);
      v16h af = frag_load16(&Aedge[rbase * K1 + k0 + koff]);
      #pragma unroll
      for (int nt = 0; nt < NTW; ++nt)
        acc[mt][nt] = __builtin_amdgcn_wmma_f32_16x16x32_f16(
            false, af, false, bf[nt], (short)0, acc[mt][nt], false, false);
    }
  }
  // epilogue -> H1 (bias + BN + ReLU, f16)
  #pragma unroll
  for (int mt = 0; mt < 2; ++mt) {
    #pragma unroll
    for (int nt = 0; nt < NTW; ++nt) {
      int col = nw * (COUT / 4) + nt * 16 + (lane & 15);
      float bv = b1[col];
      float sc = bng[col] * rsqrtf(1.f + EPSBN);
      float sh = bnb[col];
      #pragma unroll
      for (int v = 0; v < 8; ++v) {
        int row = mw * 32 + mt * 16 + v + ((lane >= 16) ? 8 : 0);
        float y = (acc[mt][nt][v] + bv) * sc + sh;
        y = fmaxf(y, 0.f);
        H1[row * COUT + col] = (f16)y;
      }
    }
  }
  __syncthreads();

  // Phase 3: l2 = H1 @ W2
  #pragma unroll
  for (int i = 0; i < 2; ++i)
    #pragma unroll
    for (int j = 0; j < NTW; ++j)
      #pragma unroll
      for (int e = 0; e < 8; ++e) acc[i][j][e] = 0.f;

  for (int k0 = 0; k0 < COUT; k0 += 32) {
    v16h bf[NTW];
    #pragma unroll
    for (int nt = 0; nt < NTW; ++nt) {
      int colBase = nw * (COUT / 4) + nt * 16;
      bf[nt] = frag_load16(W2t + (long)(colBase + (lane & 15)) * COUT + k0 + koff);
    }
    #pragma unroll
    for (int mt = 0; mt < 2; ++mt) {
      int rbase = mw * 32 + mt * 16 + (lane & 15);
      v16h af = frag_load16(&H1[rbase * COUT + k0 + koff]);
      #pragma unroll
      for (int nt = 0; nt < NTW; ++nt)
        acc[mt][nt] = __builtin_amdgcn_wmma_f32_16x16x32_f16(
            false, af, false, bf[nt], (short)0, acc[mt][nt], false, false);
    }
  }

  // Phase 4: max over 32 neighbor rows of this wave's point, + bias2
  #pragma unroll
  for (int nt = 0; nt < NTW; ++nt) {
    float r = -3.4e38f;
    #pragma unroll
    for (int v = 0; v < 8; ++v)
      r = fmaxf(r, fmaxf(acc[0][nt][v], acc[1][nt][v]));
    float o = __shfl_xor(r, 16);      // combine M halves (lanes n <-> n+16)
    r = fmaxf(r, o);
    if (lane < 16) {
      int col = nw * (COUT / 4) + nt * 16 + lane;
      Out[(iBase + mw) * ldo + ooff + col] = r + b2[col];
    }
  }
}

// ---------------------------------------------------------------------------
// Brute-force kNN (top-30 incl. self) within each cloud, LDS-tiled, float4.
// Block = 64 queries (2 waves); stores GLOBAL row indices.
// Top-k kept entirely in registers: insert-at-tail + unrolled bubble pass
// (static indices only -> no scratch spills).
// ---------------------------------------------------------------------------
template <int C>
__global__ void __launch_bounds__(64)
k_knn(const float* __restrict__ X, long ldx, long xoff,
      int* __restrict__ IDX, int nPerB) {
  __shared__ __align__(16) float Xq[64 * C];
  __shared__ __align__(16) float Xj[32 * C];
  const int tid = threadIdx.x;
  const long q = (long)blockIdx.x * 64 + tid;
  const long b = q / nPerB;
  const long jbase = b * (long)nPerB;

  {
    const float4* s = (const float4*)(X + q * ldx + xoff);
    float4* d = (float4*)&Xq[tid * C];
    for (int c4 = 0; c4 < C / 4; ++c4) d[c4] = s[c4];
  }

  float bd[30]; int bi[30];
  #pragma unroll
  for (int e = 0; e < 30; ++e) { bd[e] = 3.4e38f; bi[e] = (int)jbase; }

  float4* Xj4 = (float4*)Xj;
  for (int jt = 0; jt < nPerB; jt += 32) {
    __syncthreads();
    for (int u = 0; u < (32 * C / 4) / 64; ++u) {
      int e4 = tid + u * 64;
      int rr = e4 / (C / 4), cc = e4 % (C / 4);
      Xj4[e4] = ((const float4*)(X + (jbase + jt + rr) * ldx + xoff))[cc];
    }
    __syncthreads();
    const float4* xq4 = (const float4*)&Xq[tid * C];
    for (int jj = 0; jj < 32; ++jj) {
      const float4* xj4 = (const float4*)&Xj[jj * C];
      float d = 0.f;
      for (int c4 = 0; c4 < C / 4; ++c4) {
        float4 a = xq4[c4], bb = xj4[c4];
        float t0 = a.x - bb.x, t1 = a.y - bb.y;
        float t2 = a.z - bb.z, t3 = a.w - bb.w;
        d += t0 * t0 + t1 * t1 + t2 * t2 + t3 * t3;
      }
      if (d < bd[29]) {
        bd[29] = d; bi[29] = (int)(jbase + jt + jj);
        #pragma unroll
        for (int p = 29; p > 0; --p) {     // bubble up, static indices
          float dl = bd[p - 1], dh = bd[p];
          int   il = bi[p - 1], ih = bi[p];
          bool sw = dh < dl;
          bd[p - 1] = sw ? dh : dl;  bd[p] = sw ? dl : dh;
          bi[p - 1] = sw ? ih : il;  bi[p] = sw ? il : ih;
        }
      }
    }
  }
  for (int e = 0; e < 30; ++e) IDX[q * 30 + e] = bi[e];
}

// ---------------------------------------------------------------------------
// Small helper kernels (tiny FLOP counts)
// ---------------------------------------------------------------------------
__global__ void k_rowmax(const float* __restrict__ A, int ld, int nPerB, int C,
                         int total, float* __restrict__ Out) {
  int t = blockIdx.x * 128 + threadIdx.x;
  if (t >= total) return;
  int b = t / C, c = t % C;
  float mx = -3.4e38f;
  for (int n = 0; n < nPerB; ++n)
    mx = fmaxf(mx, A[((long)b * nPerB + n) * (long)ld + c]);
  Out[(long)b * C + c] = mx;
}

__global__ void k_pool(const float* __restrict__ A, int ld, int nPerB,
                       float* __restrict__ G) {
  int b = blockIdx.x, c = threadIdx.x;    // 128 threads
  float mx = -3.4e38f, s = 0.f, ss = 0.f;
  for (int n = 0; n < nPerB; ++n) {
    float v = A[((long)b * nPerB + n) * (long)ld + c];
    mx = fmaxf(mx, v); s += v; ss += v * v;
  }
  float mean = s / nPerB;
  float var = (ss - nPerB * mean * mean) / (nPerB - 1);   // ddof=1
  G[b * 384 + c] = mx;
  G[b * 384 + 128 + c] = mean;
  G[b * 384 + 256 + c] = sqrtf(fmaxf(var, 0.f));
}

__global__ void k_fc(const float* __restrict__ A, int lda, int M, int K,
                     const float* __restrict__ W, const float* __restrict__ bias,
                     const float* __restrict__ bng, const float* __restrict__ bnb,
                     int relu, int eyeK, float* __restrict__ Out, int Nn) {
  int t = blockIdx.x * 128 + threadIdx.x;
  if (t >= M * Nn) return;
  int m = t / Nn, n = t % Nn;
  float s = bias[n];
  for (int k = 0; k < K; ++k) s += A[(long)m * lda + k] * W[(long)k * Nn + n];
  if (bng) s = s * (bng[n] * rsqrtf(1.f + EPSBN)) + bnb[n];
  if (relu) s = fmaxf(s, 0.f);
  if (eyeK && (n % (eyeK + 1) == 0)) s += 1.f;
  Out[(long)m * Nn + n] = s;
}

__global__ void k_bmm(const float* __restrict__ X, long ldx, long xoff,
                      const float* __restrict__ T, int C, int nPerB,
                      float* __restrict__ Out, long ldo, long ooff, long total) {
  long t = (long)blockIdx.x * 256 + threadIdx.x;
  if (t >= total) return;
  int co = (int)(t % C);
  long row = t / C;
  long b = row / nPerB;
  const float* x = X + row * ldx + xoff;
  const float* tm = T + b * (long)C * C;
  float s = 0.f;
  for (int ci = 0; ci < C; ++ci) s += x[ci] * tm[(long)ci * C + co];
  Out[row * ldo + ooff + co] = s;
}

// ---------------------------------------------------------------------------
// Input index map (setup_inputs flatten order)
// ---------------------------------------------------------------------------
enum {
  IN_DATA = 0,
  STN_C1W = 1, STN_C1B = 2, STN_BN1G = 3, STN_BN1B = 4,
  STN_C2W = 5, STN_C2B = 6, STN_BN2G = 7, STN_BN2B = 8,
  STN_C3W = 9, STN_C3B = 10, STN_BN3G = 11, STN_BN3B = 12,
  STN_F1W = 13, STN_F1B = 14, STN_BN4G = 15, STN_BN4B = 16,
  STN_F2W = 17, STN_F2B = 18, STN_BN5G = 19, STN_BN5B = 20,
  STN_F3W = 21, STN_F3B = 22,
  FST_C1W = 23, FST_C1B = 24, FST_BN1G = 25, FST_BN1B = 26,
  FST_C2W = 27, FST_C2B = 28, FST_BN2G = 29, FST_BN2B = 30,
  FST_C3W = 31, FST_C3B = 32, FST_BN3G = 33, FST_BN3B = 34,
  FST_F1W = 35, FST_F1B = 36, FST_BN4G = 37, FST_BN4B = 38,
  FST_F2W = 39, FST_F2B = 40, FST_BN5G = 41, FST_BN5B = 42,
  FST_F3W = 43, FST_F3B = 44,
  L1W = 45, L1B = 46, LBN1G = 47, LBN1B = 48,
  L2W = 49, L2B = 50, LBN2G = 51, LBN2B = 52,
  DG1_L1W = 53, DG1_L1B = 54, DG1_BNG = 55, DG1_BNB = 56, DG1_L2W = 57, DG1_L2B = 58,
  DG2_L1W = 59, DG2_L1B = 60, DG2_BNG = 61, DG2_BNB = 62, DG2_L2W = 63, DG2_L2B = 64,
  DG3_L1W = 65, DG3_L1B = 66, DG3_BNG = 67, DG3_BNB = 68, DG3_L2W = 69, DG3_L2B = 70,
  PF1W = 71, PF1B = 72, PFBN1G = 73, PFBN1B = 74,
  PF2W = 75, PF2B = 76, PFBN2G = 77, PFBN2B = 78,
  AGGW = 79, AGGB = 80, AGGBNG = 81, AGGBNB = 82,
  G1W = 83, G1B = 84, GBN1G = 85, GBN1B = 86,
  G2W = 87, G2B = 88, GBN2G = 89, GBN2B = 90,
  O1W = 91, O1B = 92, O2W = 93, O2B = 94, O3W = 95, O3B = 96
};

extern "C" void kernel_launch(void* const* d_in, const int* in_sizes, int n_in,
                              void* d_out, int out_size, void* d_ws, size_t ws_size,
                              hipStream_t stream) {
  (void)in_sizes; (void)n_in; (void)out_size; (void)ws_size;
  const int Bn = 8, Np = 2048;
  const long BN = (long)Bn * Np;

  char* ws = (char*)d_ws;
  size_t off = 0;
  auto alloc = [&](size_t bytes) -> char* {
    char* p = ws + off;
    off += (bytes + 255) & ~(size_t)255;
    return p;
  };
  auto fin = [&](int i) { return (const float*)d_in[i]; };

  // ---- weight prep for k_gemm: f32 [K][N] -> f16 [N128][Kp] ----
  auto cvt = [&](int widx, int K, int Nn) -> const f16* {
    int Kp  = (K + 31) & ~31;
    int Npd = (Nn + 127) & ~127;           // pad N to GEMM tile width
    f16* p = (f16*)alloc((size_t)Npd * Kp * sizeof(f16));
    long tot = (long)Npd * Kp;
    k_convert_wt<<<dim3((unsigned)((tot + 255) / 256)), dim3(256), 0, stream>>>(
        fin(widx), K, Nn, Kp, Npd, p);
    return p;
  };
  // ---- weight prep for k_edgeconv: exact [N][K] (N = COUT, no pad needed) --
  auto cvtE = [&](int widx, int K, int Nn) -> const f16* {
    int Kp = K;                            // K already multiple of 32
    f16* p = (f16*)alloc((size_t)Nn * Kp * sizeof(f16));
    long tot = (long)Nn * Kp;
    k_convert_wt<<<dim3((unsigned)((tot + 255) / 256)), dim3(256), 0, stream>>>(
        fin(widx), K, Nn, Kp, Nn, p);
    return p;
  };
  const f16* Wstn1 = cvt(STN_C1W, 3, 64);
  const f16* Wstn2 = cvt(STN_C2W, 64, 128);
  const f16* Wstn3 = cvt(STN_C3W, 128, 1024);
  const f16* Wfst1 = cvt(FST_C1W, 64, 64);
  const f16* Wfst2 = cvt(FST_C2W, 64, 128);
  const f16* Wfst3 = cvt(FST_C3W, 128, 1024);
  const f16* Wl1   = cvt(L1W, 3, 64);
  const f16* Wl2   = cvt(L2W, 64, 64);
  const f16* Wdg1a = cvtE(DG1_L1W, 128, 64);
  const f16* Wdg1b = cvtE(DG1_L2W, 64, 64);
  const f16* Wdg2a = cvtE(DG2_L1W, 128, 128);
  const f16* Wdg2b = cvtE(DG2_L2W, 128, 128);
  const f16* Wdg3a = cvtE(DG3_L1W, 256, 256);
  const f16* Wdg3b = cvtE(DG3_L2W, 256, 256);
  const f16* Wpf1  = cvt(PF1W, 512, 512);
  const f16* Wpf2  = cvt(PF2W, 512, 256);
  const f16* Wagg  = cvt(AGGW, 256, 128);

  // ---- activation buffers ----
  float* bufA  = (float*)alloc((size_t)BN * 1024 * 4);  // big ping
  float* bufB  = (float*)alloc((size_t)BN * 256 * 4);
  float* bufC  = (float*)alloc((size_t)BN * 128 * 4);
  float* bufLF = (float*)alloc((size_t)BN * 64 * 4);
  float* bufXT = (float*)alloc((size_t)BN * 3 * 4);
  float* bufCat = (float*)alloc((size_t)BN * 512 * 4);  // [lf|f1|f2|f3]
  int*   bufIdx = (int*)alloc((size_t)BN * 30 * 4);
  float* g1k  = (float*)alloc(8 * 1024 * 4);
  float* g512 = (float*)alloc(8 * 512 * 4);
  float* g256 = (float*)alloc(8 * 256 * 4);
  float* g384 = (float*)alloc(8 * 384 * 4);
  float* g128 = (float*)alloc(8 * 128 * 4);
  float* g64  = (float*)alloc(8 * 64 * 4);
  float* tr9  = (float*)alloc(8 * 16 * 4);
  float* ftm  = (float*)alloc(8 * 4096 * 4);

  auto gemm = [&](const float* A, int lda, long M, int K, const f16* Wt, int Nn,
                  int bIdx, int gIdx, int bbIdx, int relu,
                  float* Out, long ldo, long ooff) {
    dim3 grid((unsigned)((Nn + 127) / 128), (unsigned)((M + 127) / 128));
    k_gemm<<<grid, 256, 0, stream>>>(A, lda, (int)M, K, Wt, Nn, fin(bIdx),
                                     gIdx >= 0 ? fin(gIdx) : nullptr,
                                     bbIdx >= 0 ? fin(bbIdx) : nullptr,
                                     relu, Out, ldo, ooff);
  };
  auto fc = [&](const float* A, int lda, int M, int K, int wIdx, int bIdx,
                int gIdx, int bbIdx, int relu, int eyeK, float* Out, int Nn) {
    int tot = M * Nn;
    k_fc<<<dim3((unsigned)((tot + 127) / 128)), 128, 0, stream>>>(
        A, lda, M, K, fin(wIdx), fin(bIdx),
        gIdx >= 0 ? fin(gIdx) : nullptr, bbIdx >= 0 ? fin(bbIdx) : nullptr,
        relu, eyeK, Out, Nn);
  };

  const float* Xin = fin(IN_DATA);

  // ---- STN(3) ----
  gemm(Xin, 3, BN, 3, Wstn1, 64, STN_C1B, STN_BN1G, STN_BN1B, 1, bufC, 64, 0);
  gemm(bufC, 64, BN, 64, Wstn2, 128, STN_C2B, STN_BN2G, STN_BN2B, 1, bufB, 128, 0);
  gemm(bufB, 128, BN, 128, Wstn3, 1024, STN_C3B, STN_BN3G, STN_BN3B, 1, bufA, 1024, 0);
  k_rowmax<<<(8 * 1024 + 127) / 128, 128, 0, stream>>>(bufA, 1024, Np, 1024, 8 * 1024, g1k);
  fc(g1k, 1024, 8, 1024, STN_F1W, STN_F1B, STN_BN4G, STN_BN4B, 1, 0, g512, 512);
  fc(g512, 512, 8, 512, STN_F2W, STN_F2B, STN_BN5G, STN_BN5B, 1, 0, g256, 256);
  fc(g256, 256, 8, 256, STN_F3W, STN_F3B, -1, -1, 0, 3, tr9, 9);
  k_bmm<<<(unsigned)((BN * 3 + 255) / 256), 256, 0, stream>>>(
      Xin, 3, 0, tr9, 3, Np, bufXT, 3, 0, BN * 3);

  // ---- local MLP ----
  gemm(bufXT, 3, BN, 3, Wl1, 64, L1B, LBN1G, LBN1B, 1, bufC, 64, 0);
  gemm(bufC, 64, BN, 64, Wl2, 64, L2B, LBN2G, LBN2B, 1, bufLF, 64, 0);

  // ---- fSTN(64) ----
  gemm(bufLF, 64, BN, 64, Wfst1, 64, FST_C1B, FST_BN1G, FST_BN1B, 1, bufC, 64, 0);
  gemm(bufC, 64, BN, 64, Wfst2, 128, FST_C2B, FST_BN2G, FST_BN2B, 1, bufB, 128, 0);
  gemm(bufB, 128, BN, 128, Wfst3, 1024, FST_C3B, FST_BN3G, FST_BN3B, 1, bufA, 1024, 0);
  k_rowmax<<<(8 * 1024 + 127) / 128, 128, 0, stream>>>(bufA, 1024, Np, 1024, 8 * 1024, g1k);
  fc(g1k, 1024, 8, 1024, FST_F1W, FST_F1B, FST_BN4G, FST_BN4B, 1, 0, g512, 512);
  fc(g512, 512, 8, 512, FST_F2W, FST_F2B, FST_BN5G, FST_BN5B, 1, 0, g256, 256);
  fc(g256, 256, 8, 256, FST_F3W, FST_F3B, -1, -1, 0, 64, ftm, 4096);
  k_bmm<<<(unsigned)((BN * 64 + 255) / 256), 256, 0, stream>>>(
      bufLF, 64, 0, ftm, 64, Np, bufCat, 512, 0, BN * 64);     // lf -> cat[0:64]

  // ---- EdgeConv x3 (kNN + fused 2-layer MLP + max) ----
  k_knn<64><<<(unsigned)(BN / 64), 64, 0, stream>>>(bufCat, 512, 0, bufIdx, Np);
  k_edgeconv<64, 64><<<(unsigned)(BN / 2), 256, 0, stream>>>(
      bufCat, 512, 0, bufIdx, Wdg1a, fin(DG1_L1B), fin(DG1_BNG), fin(DG1_BNB),
      Wdg1b, fin(DG1_L2B), bufCat, 512, 64);                   // f1 -> cat[64:128]

  k_knn<64><<<(unsigned)(BN / 64), 64, 0, stream>>>(bufCat, 512, 64, bufIdx, Np);
  k_edgeconv<64, 128><<<(unsigned)(BN / 2), 256, 0, stream>>>(
      bufCat, 512, 64, bufIdx, Wdg2a, fin(DG2_L1B), fin(DG2_BNG), fin(DG2_BNB),
      Wdg2b, fin(DG2_L2B), bufCat, 512, 128);                  // f2 -> cat[128:256]

  k_knn<128><<<(unsigned)(BN / 64), 64, 0, stream>>>(bufCat, 512, 128, bufIdx, Np);
  k_edgeconv<128, 256><<<(unsigned)(BN / 2), 256, 0, stream>>>(
      bufCat, 512, 128, bufIdx, Wdg3a, fin(DG3_L1B), fin(DG3_BNG), fin(DG3_BNB),
      Wdg3b, fin(DG3_L2B), bufCat, 512, 256);                  // f3 -> cat[256:512]

  // ---- head ----
  gemm(bufCat, 512, BN, 512, Wpf1, 512, PF1B, PFBN1G, PFBN1B, 1, bufA, 512, 0);
  gemm(bufA, 512, BN, 512, Wpf2, 256, PF2B, PFBN2G, PFBN2B, 1, bufB, 256, 0);
  gemm(bufB, 256, BN, 256, Wagg, 128, AGGB, AGGBNG, AGGBNB, 1, bufC, 128, 0);
  k_pool<<<8, 128, 0, stream>>>(bufC, 128, Np, g384);
  fc(g384, 384, 8, 384, G1W, G1B, GBN1G, GBN1B, 1, 0, g512, 512);
  fc(g512, 512, 8, 512, G2W, G2B, GBN2G, GBN2B, 1, 0, g256, 256);
  fc(g256, 256, 8, 256, O1W, O1B, -1, -1, 1, 0, g128, 128);
  fc(g128, 128, 8, 128, O2W, O2B, -1, -1, 1, 0, g64, 64);
  fc(g64, 64, 8, 64, O3W, O3B, -1, -1, 0, 0, (float*)d_out, 55);
}
